// ProteinMPNN_49091476194160
// MI455X (gfx1250) — compile-verified, matrix operand order checked by reading, not run on hardware
//
#include <hip/hip_runtime.h>
#include <math.h>

typedef __bf16 bf16;
typedef __bf16 v16bf __attribute__((ext_vector_type(16)));
typedef float  v8f  __attribute__((ext_vector_type(8)));

#define N_NODES 10000
#define N_EDGES 300000

// ---------------------------------------------------------------------------
// helpers
// ---------------------------------------------------------------------------
__device__ __forceinline__ float gelu_exact(float x) {
  return 0.5f * x * (1.0f + erff(x * 0.70710678118654752f));
}
__device__ __forceinline__ unsigned pack2(float lo, float hi) {
  union { bf16 h[2]; unsigned u; } t;
  t.h[0] = (bf16)lo; t.h[1] = (bf16)hi;
  return t.u;
}
union Frag { unsigned u[8]; v16bf v; };

// A fragment (16x32 bf16) from LDS, row-major with even element stride.
// Per ISA 7.12.2: lane m=L&15, half=L>>4; VGPR v holds K pair at
// kbase = k0 + (v>>2)*16 + half*8 + (v&3)*2
__device__ __forceinline__ v16bf load_a_lds(const bf16* sA, int stride, int k0, int lane) {
  const int m = lane & 15, half = lane >> 4;
  const unsigned* row = (const unsigned*)(sA + m * stride);
  Frag f;
#pragma unroll
  for (int v = 0; v < 8; ++v) {
    int kb = k0 + ((v >> 2) << 4) + (half << 3) + ((v & 3) << 1);
    f.u[v] = row[kb >> 1];
  }
  return f.v;
}

// B fragment from pre-packed weights: Wp[((kt*nTiles+nt)*32+lane)*8 + v]
// (8 contiguous dwords per lane -> two global_load_b128)
__device__ __forceinline__ v16bf load_b_pk(const unsigned* __restrict__ Wp,
                                           int kt, int nt, int nTiles, int lane) {
  const unsigned* p = Wp + (((size_t)(kt * nTiles + nt)) * 32 + lane) * 8;
  Frag f;
#pragma unroll
  for (int v = 0; v < 8; ++v) f.u[v] = p[v];
  return f.v;
}

__device__ __forceinline__ v8f wmma32(v16bf a, v16bf b, v8f c) {
  return __builtin_amdgcn_wmma_f32_16x16x32_bf16(false, a, false, b, (short)0, c, false, false);
}
__device__ __forceinline__ v8f vzero8() {
  v8f z;
#pragma unroll
  for (int i = 0; i < 8; ++i) z[i] = 0.f;
  return z;
}

// 16xK @ Kx128 per block (4 waves, 2 N-tiles per wave)
template <int KIN>
__device__ __forceinline__ void gemm_16xK_128(const bf16* sAin, const unsigned* __restrict__ Wp,
                                              int lane, int wave, v8f& a0, v8f& a1) {
  a0 = vzero8(); a1 = vzero8();
  const int nt0 = wave * 2;
#pragma unroll
  for (int kt = 0; kt < KIN / 32; ++kt) {
    v16bf a = load_a_lds(sAin, KIN, kt * 32, lane);
    a0 = wmma32(a, load_b_pk(Wp, kt, nt0, 8, lane), a0);
    a1 = wmma32(a, load_b_pk(Wp, kt, nt0 + 1, 8, lane), a1);
  }
}

// ---------------------------------------------------------------------------
// weight packing: f32 [K x Nsrc] -> bf16 fragment-order, N padded, K mult of 32
// ---------------------------------------------------------------------------
__global__ void pack_weight_kernel(const float* __restrict__ W, unsigned* __restrict__ Wp,
                                   int Kdim, int N, int Nsrc, int total) {
  int tid = blockIdx.x * blockDim.x + threadIdx.x;
  if (tid >= total) return;
  int v = tid & 7, lane = (tid >> 3) & 31, rest = tid >> 8;
  int nT = N >> 4;
  int nt = rest % nT, kt = rest / nT;
  int n = nt * 16 + (lane & 15);
  int kb = kt * 32 + ((v >> 2) << 4) + ((lane >> 4) << 3) + ((v & 3) << 1);
  float lo = 0.f, hi = 0.f;
  if (n < Nsrc) { lo = W[(size_t)kb * Nsrc + n]; hi = W[(size_t)(kb + 1) * Nsrc + n]; }
  Wp[tid] = pack2(lo, hi);
  (void)Kdim;
}

// ---------------------------------------------------------------------------
// geometry: N, Ca, C, O, Cb  per node -> atoms[node][5][3]
// ---------------------------------------------------------------------------
__global__ void atoms_kernel(const float* __restrict__ x, float* __restrict__ atoms, int NN) {
  int i = blockIdx.x * blockDim.x + threadIdx.x;
  if (i >= NN) return;
  const float* p = x + (size_t)i * 12;
  float Nx = p[0], Ny = p[1], Nz = p[2];
  float Cax = p[3], Cay = p[4], Caz = p[5];
  float Cx = p[6], Cy = p[7], Cz = p[8];
  float Ox = p[9], Oy = p[10], Oz = p[11];
  float bx = Cax - Nx, by = Cay - Ny, bz = Caz - Nz;
  float cx = Cx - Cax, cy = Cy - Cay, cz = Cz - Caz;
  float ax = by * cz - bz * cy, ay = bz * cx - bx * cz, az = bx * cy - by * cx;
  float Cbx = -0.58273431f * ax + 0.56802827f * bx - 0.54067466f * cx + Cax;
  float Cby = -0.58273431f * ay + 0.56802827f * by - 0.54067466f * cy + Cay;
  float Cbz = -0.58273431f * az + 0.56802827f * bz - 0.54067466f * cz + Caz;
  float* o = atoms + (size_t)i * 15;
  o[0] = Nx;  o[1] = Ny;  o[2] = Nz;
  o[3] = Cax; o[4] = Cay; o[5] = Caz;
  o[6] = Cx;  o[7] = Cy;  o[8] = Cz;
  o[9] = Ox;  o[10] = Oy; o[11] = Oz;
  o[12] = Cbx; o[13] = Cby; o[14] = Cbz;
}

__global__ void zero_u32_kernel(unsigned* __restrict__ p, long long n) {
  long long i = (long long)blockIdx.x * blockDim.x + threadIdx.x;
  if (i < n) p[i] = 0u;
}

__global__ void label_kernel(const int* __restrict__ seq, const float* __restrict__ emb,
                             bf16* __restrict__ out, int NN) {
  int i = blockIdx.x * blockDim.x + threadIdx.x;
  if (i >= NN * 128) return;
  int node = i >> 7, c = i & 127;
  out[i] = (bf16)emb[(size_t)seq[node] * 128 + c];
}

// ---------------------------------------------------------------------------
// h_v = LN(h_v + add*scale) ; one node per wave32
// ---------------------------------------------------------------------------
__global__ void __launch_bounds__(256)
node_ln_kernel(float* __restrict__ hv, bf16* __restrict__ hvB, float* __restrict__ add,
               float scale, const float* __restrict__ g, const float* __restrict__ b,
               int zeroAdd, int NN) {
  int wave = threadIdx.x >> 5, lane = threadIdx.x & 31;
  int node = blockIdx.x * 8 + wave;
  if (node >= NN) return;
  float v[4]; float s = 0.f, ss = 0.f;
#pragma unroll
  for (int i = 0; i < 4; ++i) {
    int c = lane + 32 * i;
    size_t idx = (size_t)node * 128 + c;
    float a = add[idx];
    v[i] = hv[idx] + a * scale;
    if (zeroAdd) add[idx] = 0.f;
    s += v[i]; ss += v[i] * v[i];
  }
#pragma unroll
  for (int off = 16; off > 0; off >>= 1) {
    s += __shfl_xor(s, off, 32);
    ss += __shfl_xor(ss, off, 32);
  }
  float mu = s * (1.0f / 128.0f);
  float var = ss * (1.0f / 128.0f) - mu * mu;
  float rs = rsqrtf(var + 1e-5f);
#pragma unroll
  for (int i = 0; i < 4; ++i) {
    int c = lane + 32 * i;
    size_t idx = (size_t)node * 128 + c;
    float o = (v[i] - mu) * rs * g[c] + b[c];
    hv[idx] = o;
    hvB[idx] = (bf16)o;
  }
}

// ---------------------------------------------------------------------------
// generic dense GEMM: out[M x N] = act(A[M x K](bf16) @ Wp + bias)
// block = 128 threads, 16 rows; wave handles NTPW n-tiles
// ---------------------------------------------------------------------------
template <int NTPW, int ACT>
__global__ void __launch_bounds__(128)
mat_gemm_kernel(const bf16* __restrict__ A, int lda,
                const unsigned* __restrict__ Wp, const float* __restrict__ bias,
                int Kdim, int N, int Nsrc,
                float* __restrict__ outF, bf16* __restrict__ outB, int ldo) {
  __shared__ bf16 sA[16 * 512];
  const int t = threadIdx.x, lane = t & 31, wave = t >> 5;
  const int m0 = blockIdx.x * 16;
  const int dw = Kdim >> 1;
  for (int i = t; i < 16 * dw; i += 128) {
    int m = i / dw, c = i - m * dw;
    ((unsigned*)sA)[m * dw + c] = ((const unsigned*)(A + (size_t)(m0 + m) * lda))[c];
  }
  __syncthreads();
  const int nTiles = N >> 4;
  v8f acc[NTPW];
#pragma unroll
  for (int j = 0; j < NTPW; ++j) acc[j] = vzero8();
  const int kts = Kdim >> 5;
  for (int kt = 0; kt < kts; ++kt) {
    v16bf a = load_a_lds(sA, Kdim, kt << 5, lane);
#pragma unroll
    for (int j = 0; j < NTPW; ++j) {
      int nt = wave * NTPW + j;
      if (nt < nTiles) acc[j] = wmma32(a, load_b_pk(Wp, kt, nt, nTiles, lane), acc[j]);
    }
  }
  const int n = lane & 15, half = lane >> 4;
#pragma unroll
  for (int j = 0; j < NTPW; ++j) {
    int nt = wave * NTPW + j;
    if (nt >= nTiles) continue;
    int ncol = nt * 16 + n;
    float bb = (ncol < Nsrc) ? bias[ncol] : 0.f;
#pragma unroll
    for (int r = 0; r < 8; ++r) {
      int m = r + (half << 3);
      float v = acc[j][r] + bb;
      if (ACT == 1) v = gelu_exact(v);
      if (ncol < Nsrc) {
        size_t idx = (size_t)(m0 + m) * ldo + ncol;
        if (outF) outF[idx] = v;
        if (outB) outB[idx] = (bf16)v;
      }
    }
  }
}

// ---------------------------------------------------------------------------
// fused edge features (25 dists x 16 RBF + pos emb) -> l1 -> LN -> l2 -> h_e
// 16 edges per block, 128 threads (4 waves)
// ---------------------------------------------------------------------------
__global__ void __launch_bounds__(128)
edge_init_kernel(const int* __restrict__ eidx, const float* __restrict__ atoms,
                 const int* __restrict__ residue, const int* __restrict__ chain,
                 const float* __restrict__ pos_emb,
                 const unsigned* __restrict__ W1p, const float* __restrict__ b1,
                 const unsigned* __restrict__ W2p, const float* __restrict__ b2,
                 const float* __restrict__ lng, const float* __restrict__ lnb,
                 float* __restrict__ heF, bf16* __restrict__ heB, int E) {
  __shared__ bf16 sA[16 * 416];
  __shared__ float sD[16 * 25];
  __shared__ float sOut[16 * 128];
  __shared__ bf16 sMid[16 * 128];
  __shared__ float sRed[128], sRed2[128];
  __shared__ float sMu[16], sRs[16];
  __shared__ int sRow[16], sCol[16];
  const int t = threadIdx.x, lane = t & 31, wave = t >> 5;
  const int e0 = blockIdx.x * 16;
  if (t < 16) { int e = e0 + t; sRow[t] = eidx[e]; sCol[t] = eidx[E + e]; }
  __syncthreads();
  // 25 pairwise distances per edge
  for (int i = t; i < 400; i += 128) {
    int m = i / 25, p = i - m * 25, ai = p / 5, bj = p - ai * 5;
    const float* pa = atoms + (size_t)sRow[m] * 15 + ai * 3;
    const float* pb = atoms + (size_t)sCol[m] * 15 + bj * 3;
    float dx = pa[0] - pb[0], dy = pa[1] - pb[1], dz = pa[2] - pb[2];
    sD[m * 25 + p] = sqrtf(dx * dx + dy * dy + dz * dz + 1e-6f);
  }
  __syncthreads();
  // 416 features per edge (RBF + pos emb), as bf16 into LDS A tile
  for (int i = t; i < 16 * 416; i += 128) {
    int m = i / 416, f = i - m * 416;
    float val;
    if (f < 400) {
      int p = f >> 4, k = f & 15;
      float mu = 2.0f + (float)k * (20.0f / 15.0f);
      float z = (sD[m * 25 + p] - mu) * 0.8f;  // / sigma, sigma = 1.25
      val = expf(-z * z);
    } else {
      int k = f - 400;
      int r = sRow[m], c = sCol[m];
      int d;
      if (chain[r] == chain[c]) {
        d = (residue[r] - residue[c]) + 32;
        d = d < 0 ? 0 : (d > 64 ? 64 : d);
      } else d = 65;
      val = pos_emb[d * 16 + k];
    }
    sA[m * 416 + f] = (bf16)val;
  }
  __syncthreads();
  // GEMM1: [16x416] @ [416x128] + b1
  {
    v8f a0, a1;
    gemm_16xK_128<416>(sA, W1p, lane, wave, a0, a1);
    int n = lane & 15, half = lane >> 4, nt0 = wave * 2;
#pragma unroll
    for (int j = 0; j < 2; ++j) {
      int ncol = (nt0 + j) * 16 + n;
      float bb = b1[ncol];
      v8f& A = j ? a1 : a0;
#pragma unroll
      for (int r = 0; r < 8; ++r) sOut[(r + (half << 3)) * 128 + ncol] = A[r] + bb;
    }
  }
  __syncthreads();
  // LayerNorm across 128 per edge row
  {
    int m = t >> 3, seg = t & 7;
    float s = 0.f, ss = 0.f;
    for (int c = seg * 16; c < seg * 16 + 16; ++c) {
      float v = sOut[m * 128 + c];
      s += v; ss += v * v;
    }
    sRed[t] = s; sRed2[t] = ss;
  }
  __syncthreads();
  if (t < 16) {
    float s = 0.f, ss = 0.f;
    for (int g2 = 0; g2 < 8; ++g2) { s += sRed[t * 8 + g2]; ss += sRed2[t * 8 + g2]; }
    float mu = s * (1.0f / 128.0f);
    float var = ss * (1.0f / 128.0f) - mu * mu;
    sMu[t] = mu; sRs[t] = rsqrtf(var + 1e-5f);
  }
  __syncthreads();
  for (int i = t; i < 2048; i += 128) {
    int m = i >> 7, c = i & 127;
    float v = (sOut[i] - sMu[m]) * sRs[m] * lng[c] + lnb[c];
    sMid[i] = (bf16)v;
  }
  __syncthreads();
  // GEMM2: [16x128] @ [128x128] + b2 -> h_e (f32 + bf16)
  {
    v8f a0, a1;
    gemm_16xK_128<128>(sMid, W2p, lane, wave, a0, a1);
    int n = lane & 15, half = lane >> 4, nt0 = wave * 2;
#pragma unroll
    for (int j = 0; j < 2; ++j) {
      int ncol = (nt0 + j) * 16 + n;
      float bb = b2[ncol];
      v8f& A = j ? a1 : a0;
#pragma unroll
      for (int r = 0; r < 8; ++r) {
        int m = r + (half << 3);
        float v = A[r] + bb;
        size_t idx = (size_t)(e0 + m) * 128 + ncol;
        heF[idx] = v;
        heB[idx] = (bf16)v;
      }
    }
  }
}

// ---------------------------------------------------------------------------
// fused edge MLP3 over 16-edge tiles.
// MODE 0: enc message  A=[hv[col] hv[row] h_e]          -> atomicAdd agg[col]
// MODE 1: enc edge-upd A=[hv[row] hv[col] h_e]          -> h_e = LN(h_e + m)
// MODE 2: dec message  A=[hv[col] hv[row] h_e mask*lbl] -> atomicAdd agg[col]
// ---------------------------------------------------------------------------
template <int KIN, int MODE>
__global__ void __launch_bounds__(128)
edge_mlp3_kernel(const int* __restrict__ eidx,
                 const bf16* __restrict__ hvB, const bf16* __restrict__ heBin,
                 const bf16* __restrict__ lblB, const float* __restrict__ emask,
                 const unsigned* __restrict__ W1p, const float* __restrict__ b1,
                 const unsigned* __restrict__ W2p, const float* __restrict__ b2,
                 const unsigned* __restrict__ W3p, const float* __restrict__ b3,
                 float* __restrict__ agg,
                 float* __restrict__ heF, bf16* __restrict__ heBout,
                 const float* __restrict__ lng, const float* __restrict__ lnb,
                 int E) {
  __shared__ bf16 sA[16 * KIN];
  __shared__ bf16 sMid[16 * 128];
  __shared__ bf16 sB[16 * 128];
  __shared__ float sOut[16 * 128];
  __shared__ float sRed[128], sRed2[128];
  __shared__ float sMu[16], sRs[16];
  __shared__ int sRow[16], sCol[16];
  __shared__ float sMask[16];
  const int t = threadIdx.x, lane = t & 31, wave = t >> 5;
  const int e0 = blockIdx.x * 16;
  if (t < 16) {
    int e = e0 + t;
    sRow[t] = eidx[e];
    sCol[t] = eidx[E + e];
    if (MODE == 2) sMask[t] = emask[e];
  }
  __syncthreads();
  // gather concat input tile (dword = 2 packed bf16)
  const int dw = KIN / 2;
  for (int i = t; i < 16 * dw; i += 128) {
    int m = i / dw, c2 = i - m * dw;
    int blk = c2 >> 6, off = c2 & 63;
    int e = e0 + m;
    unsigned val;
    if (MODE == 1) {
      const bf16* src = (blk == 0) ? hvB + (size_t)sRow[m] * 128
                      : (blk == 1) ? hvB + (size_t)sCol[m] * 128
                                   : heBin + (size_t)e * 128;
      val = ((const unsigned*)src)[off];
    } else {
      if (blk == 0)      val = ((const unsigned*)(hvB + (size_t)sCol[m] * 128))[off];
      else if (blk == 1) val = ((const unsigned*)(hvB + (size_t)sRow[m] * 128))[off];
      else if (blk == 2) val = ((const unsigned*)(heBin + (size_t)e * 128))[off];
      else {
        unsigned u = ((const unsigned*)(lblB + (size_t)sRow[m] * 128))[off];
        union { bf16 h[2]; unsigned uu; } q; q.uu = u;
        float mk = sMask[m];
        q.h[0] = (bf16)((float)q.h[0] * mk);
        q.h[1] = (bf16)((float)q.h[1] * mk);
        val = q.uu;
      }
    }
    ((unsigned*)sA)[i] = val;
  }
  __syncthreads();
  const int n = lane & 15, half = lane >> 4, nt0 = wave * 2;
  // layer 1 (gelu)
  {
    v8f a0, a1;
    gemm_16xK_128<KIN>(sA, W1p, lane, wave, a0, a1);
#pragma unroll
    for (int j = 0; j < 2; ++j) {
      int ncol = (nt0 + j) * 16 + n;
      float bb = b1[ncol];
      v8f& A = j ? a1 : a0;
#pragma unroll
      for (int r = 0; r < 8; ++r)
        sMid[(r + (half << 3)) * 128 + ncol] = (bf16)gelu_exact(A[r] + bb);
    }
  }
  __syncthreads();
  // layer 2 (gelu)
  {
    v8f a0, a1;
    gemm_16xK_128<128>(sMid, W2p, lane, wave, a0, a1);
#pragma unroll
    for (int j = 0; j < 2; ++j) {
      int ncol = (nt0 + j) * 16 + n;
      float bb = b2[ncol];
      v8f& A = j ? a1 : a0;
#pragma unroll
      for (int r = 0; r < 8; ++r)
        sB[(r + (half << 3)) * 128 + ncol] = (bf16)gelu_exact(A[r] + bb);
    }
  }
  __syncthreads();
  // layer 3 (linear) + mode epilogue
  {
    v8f a0, a1;
    gemm_16xK_128<128>(sB, W3p, lane, wave, a0, a1);
    if (MODE == 0 || MODE == 2) {
#pragma unroll
      for (int j = 0; j < 2; ++j) {
        int ncol = (nt0 + j) * 16 + n;
        float bb = b3[ncol];
        v8f& A = j ? a1 : a0;
#pragma unroll
        for (int r = 0; r < 8; ++r) {
          int m = r + (half << 3);
          atomicAdd(agg + (size_t)sCol[m] * 128 + ncol, A[r] + bb);
        }
      }
    } else {
#pragma unroll
      for (int j = 0; j < 2; ++j) {
        int ncol = (nt0 + j) * 16 + n;
        float bb = b3[ncol];
        v8f& A = j ? a1 : a0;
#pragma unroll
        for (int r = 0; r < 8; ++r) sOut[(r + (half << 3)) * 128 + ncol] = A[r] + bb;
      }
      __syncthreads();
      // h_e = LN(h_e + m)
      {
        int m = t >> 3, seg = t & 7;
        float s = 0.f, ss = 0.f;
        for (int c = seg * 16; c < seg * 16 + 16; ++c) {
          float v = heF[(size_t)(e0 + m) * 128 + c] + sOut[m * 128 + c];
          s += v; ss += v * v;
        }
        sRed[t] = s; sRed2[t] = ss;
      }
      __syncthreads();
      if (t < 16) {
        float s = 0.f, ss = 0.f;
        for (int g2 = 0; g2 < 8; ++g2) { s += sRed[t * 8 + g2]; ss += sRed2[t * 8 + g2]; }
        float mu = s * (1.0f / 128.0f);
        float var = ss * (1.0f / 128.0f) - mu * mu;
        sMu[t] = mu; sRs[t] = rsqrtf(var + 1e-5f);
      }
      __syncthreads();
      for (int i = t; i < 2048; i += 128) {
        int m = i >> 7, c = i & 127;
        size_t idx = (size_t)(e0 + m) * 128 + c;
        float v = heF[idx] + sOut[i];
        float o = (v - sMu[m]) * sRs[m] * lng[c] + lnb[c];
        heF[idx] = o;
        heBout[idx] = (bf16)o;
      }
    }
  }
}

// ---------------------------------------------------------------------------
// host orchestration
// ---------------------------------------------------------------------------
extern "C" void kernel_launch(void* const* d_in, const int* in_sizes, int n_in,
                              void* d_out, int out_size, void* d_ws, size_t ws_size,
                              hipStream_t stream) {
  // Input index map: top-level insertion order, params tree-flattened
  // (dict keys sorted: dec < edge_mlp < enc < label_emb < out < pos_emb).
  const int P0 = 6;         // dec[0] base; each dec layer = 14 leaves
  const int EM = P0 + 42;   // edge_mlp: l1W,l1b,l2W,l2b,lng,lnb
  const int ENC = EM + 6;   // each enc layer = 16 leaves
  const int LBL = ENC + 48; // label_emb
  const int OUTW = LBL + 1, OUTB = LBL + 2, POS = LBL + 3;

  auto F = [&](int i) { return (const float*)d_in[i]; };
  auto I = [&](int i) { return (const int*)d_in[i]; };

  char* ws = (char*)d_ws;
  size_t off = 0;
  auto alloc = [&](size_t bytes) -> char* {
    char* p = ws + off;
    off = (off + bytes + 255) & ~(size_t)255;
    return p;
  };
  float* atoms  = (float*)alloc((size_t)N_NODES * 15 * 4);
  float* heF    = (float*)alloc((size_t)N_EDGES * 128 * 4);
  bf16*  heB    = (bf16*)alloc((size_t)N_EDGES * 128 * 2);
  float* hvF    = (float*)alloc((size_t)N_NODES * 128 * 4);
  bf16*  hvB    = (bf16*)alloc((size_t)N_NODES * 128 * 2);
  float* agg    = (float*)alloc((size_t)N_NODES * 128 * 4);
  float* ffnOut = (float*)alloc((size_t)N_NODES * 128 * 4);
  bf16*  hidden = (bf16*)alloc((size_t)N_NODES * 512 * 2);
  bf16*  lblB   = (bf16*)alloc((size_t)N_NODES * 128 * 2);

  auto packW = [&](int idx, int K, int N, int Nsrc) -> const unsigned* {
    unsigned* dst = (unsigned*)alloc((size_t)K * N * 2);
    int total = K * N / 2;
    pack_weight_kernel<<<(total + 255) / 256, 256, 0, stream>>>(F(idx), dst, K, N, Nsrc, total);
    return dst;
  };

  const unsigned* l1W = packW(EM + 0, 416, 128, 128);
  const unsigned* l2W = packW(EM + 2, 128, 128, 128);
  struct LW { const unsigned *m1, *m2, *m3, *f1, *f2; };
  LW enc[3], dec[3];
  for (int l = 0; l < 3; ++l) {
    int b = ENC + 16 * l;
    enc[l].f1 = packW(b + 0, 128, 512, 512);
    enc[l].f2 = packW(b + 2, 512, 128, 128);
    enc[l].m1 = packW(b + 10, 384, 128, 128);
    enc[l].m2 = packW(b + 12, 128, 128, 128);
    enc[l].m3 = packW(b + 14, 128, 128, 128);
  }
  for (int l = 0; l < 3; ++l) {
    int b = P0 + 14 * l;
    dec[l].f1 = packW(b + 0, 128, 512, 512);
    dec[l].f2 = packW(b + 2, 512, 128, 128);
    dec[l].m1 = packW(b + 8, 512, 128, 128);
    dec[l].m2 = packW(b + 10, 128, 128, 128);
    dec[l].m3 = packW(b + 12, 128, 128, 128);
  }
  const unsigned* outWp = packW(OUTW, 128, 32, 21);

  atoms_kernel<<<(N_NODES + 255) / 256, 256, 0, stream>>>(F(0), atoms, N_NODES);

  long long nz = (long long)N_NODES * 128;
  zero_u32_kernel<<<(int)((nz + 255) / 256), 256, 0, stream>>>((unsigned*)hvF, nz);
  zero_u32_kernel<<<(int)((nz / 2 + 255) / 256), 256, 0, stream>>>((unsigned*)hvB, nz / 2);
  zero_u32_kernel<<<(int)((nz + 255) / 256), 256, 0, stream>>>((unsigned*)agg, nz);

  label_kernel<<<(N_NODES * 128 + 255) / 256, 256, 0, stream>>>(I(2), F(LBL), lblB, N_NODES);

  edge_init_kernel<<<N_EDGES / 16, 128, 0, stream>>>(
      I(1), atoms, I(3), I(4), F(POS),
      l1W, F(EM + 1), l2W, F(EM + 3), F(EM + 4), F(EM + 5), heF, heB, N_EDGES);

  const int gE = N_EDGES / 16, gN = N_NODES / 16, gLN = N_NODES / 8;

  for (int l = 0; l < 3; ++l) {
    int b = ENC + 16 * l;
    edge_mlp3_kernel<384, 0><<<gE, 128, 0, stream>>>(
        I(1), hvB, heB, nullptr, nullptr,
        enc[l].m1, F(b + 11), enc[l].m2, F(b + 13), enc[l].m3, F(b + 15),
        agg, nullptr, nullptr, nullptr, nullptr, N_EDGES);
    node_ln_kernel<<<gLN, 256, 0, stream>>>(hvF, hvB, agg, 1.0f / 30.0f, F(b + 4), F(b + 5), 1, N_NODES);
    mat_gemm_kernel<8, 1><<<gN, 128, 0, stream>>>(hvB, 128, enc[l].f1, F(b + 1), 128, 512, 512,
                                                  nullptr, hidden, 512);
    mat_gemm_kernel<2, 0><<<gN, 128, 0, stream>>>(hidden, 512, enc[l].f2, F(b + 3), 512, 128, 128,
                                                  ffnOut, nullptr, 128);
    node_ln_kernel<<<gLN, 256, 0, stream>>>(hvF, hvB, ffnOut, 1.0f, F(b + 6), F(b + 7), 0, N_NODES);
    edge_mlp3_kernel<384, 1><<<gE, 128, 0, stream>>>(
        I(1), hvB, heB, nullptr, nullptr,
        enc[l].m1, F(b + 11), enc[l].m2, F(b + 13), enc[l].m3, F(b + 15),
        nullptr, heF, heB, F(b + 8), F(b + 9), N_EDGES);
  }

  for (int l = 0; l < 3; ++l) {
    int b = P0 + 14 * l;
    edge_mlp3_kernel<512, 2><<<gE, 128, 0, stream>>>(
        I(1), hvB, heB, lblB, F(5),
        dec[l].m1, F(b + 9), dec[l].m2, F(b + 11), dec[l].m3, F(b + 13),
        agg, nullptr, nullptr, nullptr, nullptr, N_EDGES);
    node_ln_kernel<<<gLN, 256, 0, stream>>>(hvF, hvB, agg, 1.0f / 30.0f, F(b + 4), F(b + 5), 1, N_NODES);
    mat_gemm_kernel<8, 1><<<gN, 128, 0, stream>>>(hvB, 128, dec[l].f1, F(b + 1), 128, 512, 512,
                                                  nullptr, hidden, 512);
    mat_gemm_kernel<2, 0><<<gN, 128, 0, stream>>>(hidden, 512, dec[l].f2, F(b + 3), 512, 128, 128,
                                                  ffnOut, nullptr, 128);
    node_ln_kernel<<<gLN, 256, 0, stream>>>(hvF, hvB, ffnOut, 1.0f, F(b + 6), F(b + 7), 0, N_NODES);
  }

  // logits = h_v @ Wout + bout  (N padded 21 -> 32)
  mat_gemm_kernel<1, 0><<<gN, 128, 0, stream>>>(hvB, 128, outWp, F(OUTB), 128, 32, 21,
                                                (float*)d_out, nullptr, 21);

  (void)in_sizes; (void)n_in; (void)out_size; (void)ws_size;
}
